// QuantumDiffusionDreamer_25185688224025
// MI455X (gfx1250) — matrix operand order repaired and checked by reading, not compiled, not run on hardware
//
#include <hip/hip_runtime.h>
#include <hip/hip_bf16.h>
#include <math.h>

// ---------------------------------------------------------------------------
// QuantumDiffusionDreamer for MI455X (gfx1250, wave32, WMMA bf16)
//
// segment_sum(h[src] @ W_msg)/deg == (segment_sum(h[src] * (1/deg[dst]))) @ W_msg
// so each ODE eval is: scattered f32 atomic aggregation (bandwidth-bound)
// + one fused dual 50k x 128 x 128 WMMA GEMM with tanh epilogue.
// Weights are pre-swizzled ONCE per launch into the WMMA B-fragment register
// layout (bf16, global, L2-resident) so GEMM blocks load B fragments directly.
// ---------------------------------------------------------------------------

#define HIDN 128          // HID == FEAT == 128
#define KC   32           // K-chunk per WMMA (16x16x32 bf16)
#define BM   128          // rows per GEMM block (8 waves x 16 rows)
#define NGRAPH 64

#define KA_STRIDE 40      // A-row stride in shorts (80B: 16B aligned, bank-spread)

typedef __attribute__((ext_vector_type(16))) __bf16 v16bf;
typedef __attribute__((ext_vector_type(8)))  __bf16 v8bf;
typedef __attribute__((ext_vector_type(8)))  float  v8f;

__device__ __forceinline__ unsigned short f2bf(float x) {
  union { float f; unsigned int u; } c; c.f = x;
  unsigned int u = c.u;
  u += 0x7FFFu + ((u >> 16) & 1u);   // round-to-nearest-even
  return (unsigned short)(u >> 16);
}
__device__ __forceinline__ float gelu_tanh(float x) {
  float x3 = x * x * x;
  return 0.5f * x * (1.0f + tanhf(0.7978845608028654f * (x + 0.044715f * x3)));
}
__device__ __forceinline__ float wave_allreduce_sum(float v) {
#pragma unroll
  for (int off = 16; off > 0; off >>= 1) v += __shfl_xor(v, off, 32);
  return v;
}

// ---------------------------------------------------------------------------
// One-time weight swizzle: f32 [128,128] (optionally summed with W2, for
// W_eff = W_int_top + W_int_bot) -> bf16 in B-fragment register order:
//   out[((kc4*8 + nt)*32 + L)*16 + e] = W[(kc4*32 + 8*(L>>4) + (e&7) + 2*(e&8))*128 + nt*16 + (L&15)]
// ---------------------------------------------------------------------------
__global__ void weight_swizzle_kernel(const float* __restrict__ W,
                                      const float* __restrict__ W2,
                                      unsigned short* __restrict__ o)
{
  int i = blockIdx.x * blockDim.x + threadIdx.x;
  if (i >= HIDN * HIDN) return;
  int e   = i & 15;
  int L   = (i >> 4) & 31;
  int nt  = (i >> 9) & 7;
  int kc4 = i >> 12;
  int k = kc4 * KC + (L >> 4) * 8 + (e & 7) + ((e & 8) ? 16 : 0);
  int n = nt * 16 + (L & 15);
  float v = W[k * HIDN + n];
  if (W2) v += W2[k * HIDN + n];
  o[i] = f2bf(v);
}

// ---------------------------------------------------------------------------
// WMMA GEMM: out[M,128] = act( A0@W0 (+ A1@W1) + bias )
// A: [M,128] f32 row-major, Ws: bf16 pre-swizzled B fragments (see above).
// 256 threads = 8 waves; wave w computes a 16x128 strip as 8 acc tiles.
// A fragments: ISA 7.12.2 16-bit A 16x32 layout, two 16B LDS vector loads.
// B fragments: one contiguous 32B global load per lane (L2-resident weights).
// ---------------------------------------------------------------------------
template <bool DUAL, bool TANH>
__global__ __launch_bounds__(256)
void gemm_kernel(const float* __restrict__ A0, const unsigned short* __restrict__ Ws0,
                 const float* __restrict__ A1, const unsigned short* __restrict__ Ws1,
                 const float* __restrict__ bias, float* __restrict__ out, int M)
{
  __shared__ unsigned short sA0[BM * KA_STRIDE];                     // 10 KB
  __shared__ unsigned short sA1[DUAL ? BM * KA_STRIDE : 1];

  const int tid  = threadIdx.x;
  const int wave = tid >> 5;
  const int lane = tid & 31;
  const int row0 = blockIdx.x * BM;
  const bool full = (row0 + BM) <= M;     // uniform: all rows valid

  v8f acc[8];
#pragma unroll
  for (int t = 0; t < 8; ++t)
#pragma unroll
    for (int r = 0; r < 8; ++r) acc[t][r] = 0.0f;

  const int mrow  = wave * 16 + (lane & 15);  // A-frag row within block
  const int khalf = (lane >> 4) * 8;          // A lane K-split (0 or 8)

  for (int kc = 0; kc < HIDN; kc += KC) {
    // ---- stage A chunks (BM x KC) as bf16: float4 load -> ushort4 store ----
    if (full) {
#pragma unroll
      for (int i = tid; i < BM * KC / 4; i += 256) {   // 4 iters/thread
        int r  = i >> 3;
        int k4 = (i & 7) * 4;
        size_t go = (size_t)(row0 + r) * HIDN + kc + k4;
        float4 v0 = *(const float4*)(A0 + go);
        ushort4 p0;
        p0.x = f2bf(v0.x); p0.y = f2bf(v0.y); p0.z = f2bf(v0.z); p0.w = f2bf(v0.w);
        *(ushort4*)&sA0[r * KA_STRIDE + k4] = p0;
        if (DUAL) {
          float4 v1 = *(const float4*)(A1 + go);
          ushort4 p1;
          p1.x = f2bf(v1.x); p1.y = f2bf(v1.y); p1.z = f2bf(v1.z); p1.w = f2bf(v1.w);
          *(ushort4*)&sA1[r * KA_STRIDE + k4] = p1;
        }
      }
    } else {
#pragma unroll
      for (int i = tid; i < BM * KC / 4; i += 256) {
        int r  = i >> 3;
        int k4 = (i & 7) * 4;
        int gr = row0 + r;
        float4 v0 = make_float4(0.f, 0.f, 0.f, 0.f);
        if (gr < M) v0 = *(const float4*)(A0 + (size_t)gr * HIDN + kc + k4);
        ushort4 p0;
        p0.x = f2bf(v0.x); p0.y = f2bf(v0.y); p0.z = f2bf(v0.z); p0.w = f2bf(v0.w);
        *(ushort4*)&sA0[r * KA_STRIDE + k4] = p0;
        if (DUAL) {
          float4 v1 = make_float4(0.f, 0.f, 0.f, 0.f);
          if (gr < M) v1 = *(const float4*)(A1 + (size_t)gr * HIDN + kc + k4);
          ushort4 p1;
          p1.x = f2bf(v1.x); p1.y = f2bf(v1.y); p1.z = f2bf(v1.z); p1.w = f2bf(v1.w);
          *(ushort4*)&sA1[r * KA_STRIDE + k4] = p1;
        }
      }
    }
    __syncthreads();

    // ---- A fragments: two aligned 16B vector loads + concat ----
    const v8bf* rA0 = (const v8bf*)(sA0 + mrow * KA_STRIDE + khalf);
    v8bf a0lo = rA0[0];          // K = khalf .. khalf+7
    v8bf a0hi = rA0[2];          // K = khalf+16 .. khalf+23 (16 shorts ahead)
    v16bf a0 = __builtin_shufflevector(a0lo, a0hi,
        0, 1, 2, 3, 4, 5, 6, 7, 8, 9, 10, 11, 12, 13, 14, 15);
    v16bf a1;
    if (DUAL) {
      const v8bf* rA1 = (const v8bf*)(sA1 + mrow * KA_STRIDE + khalf);
      v8bf a1lo = rA1[0];
      v8bf a1hi = rA1[2];
      a1 = __builtin_shufflevector(a1lo, a1hi,
          0, 1, 2, 3, 4, 5, 6, 7, 8, 9, 10, 11, 12, 13, 14, 15);
    }

    // ---- B fragments straight from pre-swizzled global weights ----
    const int wbase = (kc >> 5) * 8;     // chunk index * 8 tiles
#pragma unroll
    for (int nt = 0; nt < 8; ++nt) {
      v16bf b0 = *(const v16bf*)(Ws0 + (((wbase + nt) * 32) + lane) * 16);
      acc[nt] = __builtin_amdgcn_wmma_f32_16x16x32_bf16(
          false, a0, false, b0, (short)0, acc[nt], false, false);
      if (DUAL) {
        v16bf b1 = *(const v16bf*)(Ws1 + (((wbase + nt) * 32) + lane) * 16);
        acc[nt] = __builtin_amdgcn_wmma_f32_16x16x32_bf16(
            false, a1, false, b1, (short)0, acc[nt], false, false);
      }
    }
    __syncthreads();
  }

  // ---- epilogue: C/D layout (lane%16 = N, vgpr r -> M = r + 8*(lane/16)) ----
  const int nbase = lane & 15;
  const int mhalf = (lane >> 4) * 8;
#pragma unroll
  for (int nt = 0; nt < 8; ++nt) {
    int n = nt * 16 + nbase;
    float bv = bias[n];
#pragma unroll
    for (int r = 0; r < 8; ++r) {
      int m = row0 + wave * 16 + mhalf + r;
      if (full || m < M) {
        float v = acc[nt][r] + bv;
        if (TANH) v = tanhf(v);
        out[(size_t)m * HIDN + n] = v;
      }
    }
  }
}

// ---------------------------------------------------------------------------
// h = s0*gelu(LN(y0)*g0+be0) + s1*gelu(LN(y1)*g1+be1)   (bias already in y)
// ---------------------------------------------------------------------------
__global__ __launch_bounds__(256)
void ln_gelu_combine_kernel(const float* __restrict__ y0, const float* __restrict__ y1,
                            const float* __restrict__ g0, const float* __restrict__ be0,
                            const float* __restrict__ g1, const float* __restrict__ be1,
                            const float* __restrict__ alpha_p, const float* __restrict__ beta_p,
                            float* __restrict__ h, int M)
{
  int row  = blockIdx.x * 8 + (threadIdx.x >> 5);
  int lane = threadIdx.x & 31;
  if (row >= M) return;

  float a = alpha_p[0], b = beta_p[0];
  float inv = rsqrtf(a * a + b * b);
  float s0 = fabsf(a) * inv;   // sqrt(p0)
  float s1 = fabsf(b) * inv;   // sqrt(p1)

  const float* r0 = y0 + (size_t)row * HIDN + lane * 4;
  const float* r1 = y1 + (size_t)row * HIDN + lane * 4;
  float v0[4], v1[4];
  float sum0 = 0.f, sq0 = 0.f, sum1 = 0.f, sq1 = 0.f;
#pragma unroll
  for (int j = 0; j < 4; ++j) {
    v0[j] = r0[j]; v1[j] = r1[j];
    sum0 += v0[j]; sq0 += v0[j] * v0[j];
    sum1 += v1[j]; sq1 += v1[j] * v1[j];
  }
  sum0 = wave_allreduce_sum(sum0); sq0 = wave_allreduce_sum(sq0);
  sum1 = wave_allreduce_sum(sum1); sq1 = wave_allreduce_sum(sq1);
  float m0 = sum0 * (1.0f / HIDN), m1 = sum1 * (1.0f / HIDN);
  float rs0 = rsqrtf(sq0 * (1.0f / HIDN) - m0 * m0 + 1e-5f);
  float rs1 = rsqrtf(sq1 * (1.0f / HIDN) - m1 * m1 + 1e-5f);

  float* hr = h + (size_t)row * HIDN + lane * 4;
#pragma unroll
  for (int j = 0; j < 4; ++j) {
    int n = lane * 4 + j;
    float t0 = (v0[j] - m0) * rs0 * g0[n] + be0[n];
    float t1 = (v1[j] - m1) * rs1 * g1[n] + be1[n];
    hr[j] = s0 * gelu_tanh(t0) + s1 * gelu_tanh(t1);
  }
}

// ---------------------------------------------------------------------------
// h = (h + gate*tanh(LN(y)*g+be)) * cscale   (bias already in y)
// ---------------------------------------------------------------------------
__global__ __launch_bounds__(256)
void interference_kernel(const float* __restrict__ y,
                         const float* __restrict__ g, const float* __restrict__ be,
                         const float* __restrict__ alpha_p, const float* __restrict__ beta_p,
                         const float* __restrict__ phase_p,
                         float* __restrict__ h, int M)
{
  int row  = blockIdx.x * 8 + (threadIdx.x >> 5);
  int lane = threadIdx.x & 31;
  if (row >= M) return;

  float a = alpha_p[0], b = beta_p[0], ph = phase_p[0];
  float n2 = a * a + b * b;
  float p0 = a * a / n2, p1 = b * b / n2;
  float interf = 2.0f * sqrtf(p0 * p1) * cosf(ph);
  float gate = (fabsf(interf) > 0.01f) ? interf : 0.0f;
  float cscale = (p0 > p1) ? 1.0f : cosf(ph);

  const float* yr = y + (size_t)row * HIDN + lane * 4;
  float v[4];
  float sum = 0.f, sq = 0.f;
#pragma unroll
  for (int j = 0; j < 4; ++j) { v[j] = yr[j]; sum += v[j]; sq += v[j] * v[j]; }
  sum = wave_allreduce_sum(sum); sq = wave_allreduce_sum(sq);
  float m = sum * (1.0f / HIDN);
  float rs = rsqrtf(sq * (1.0f / HIDN) - m * m + 1e-5f);

  float* hr = h + (size_t)row * HIDN + lane * 4;
#pragma unroll
  for (int j = 0; j < 4; ++j) {
    int n = lane * 4 + j;
    float hi = tanhf((v[j] - m) * rs * g[n] + be[n]);
    hr[j] = (hr[j] + gate * hi) * cscale;
  }
}

// ---------------------------------------------------------------------------
// Edge machinery
// ---------------------------------------------------------------------------
__global__ void deg_kernel(const int* __restrict__ dst, float* __restrict__ deg, int E) {
  int i = blockIdx.x * blockDim.x + threadIdx.x;
  if (i < E) atomicAdd(&deg[dst[i]], 1.0f);
}
__global__ void rdeg_finalize_kernel(float* __restrict__ deg, int N) {
  int i = blockIdx.x * blockDim.x + threadIdx.x;
  if (i < N) deg[i] = 1.0f / fmaxf(deg[i], 1.0f);
}

// agg[dst] += h[src] * rdeg[dst]  — one wave per edge, lane owns 4 floats.
__global__ __launch_bounds__(256)
void edge_agg_kernel(const float* __restrict__ hin, const float* __restrict__ rdeg,
                     const int* __restrict__ src, const int* __restrict__ dst,
                     float* __restrict__ agg, int E)
{
  int wid  = blockIdx.x * 8 + (threadIdx.x >> 5);
  int lane = threadIdx.x & 31;
  int nw   = gridDim.x * 8;
  for (int e = wid; e < E; e += nw) {
    int s = src[e], d = dst[e];
    if (e + nw < E)  // CDNA5 global_prefetch of next gather row
      __builtin_prefetch(&hin[(size_t)src[e + nw] * HIDN + lane * 4], 0, 0);
    float r = rdeg[d];
    const float4 v = *(const float4*)(hin + (size_t)s * HIDN + lane * 4);
    float* ar = agg + (size_t)d * HIDN + lane * 4;
    atomicAdd(ar + 0, v.x * r);
    atomicAdd(ar + 1, v.y * r);
    atomicAdd(ar + 2, v.z * r);
    atomicAdd(ar + 3, v.w * r);
  }
}

// ---------------------------------------------------------------------------
// RK4 elementwise combine.  mode 0: acc=k; tmp=h+c*k
//                           mode 1: acc+=2k; tmp=h+c*k
//                           mode 2: h += c*(acc+k)      (c = dt/6)
// ---------------------------------------------------------------------------
__global__ void rk4_ew_kernel(float* __restrict__ h, float* __restrict__ tmp,
                              float* __restrict__ acc, const float* __restrict__ k,
                              float c, int mode, int n)
{
  int i = blockIdx.x * blockDim.x + threadIdx.x;
  if (i >= n) return;
  float kv = k[i];
  if (mode == 0)      { acc[i] = kv;          tmp[i] = h[i] + c * kv; }
  else if (mode == 1) { acc[i] += 2.0f * kv;  tmp[i] = h[i] + c * kv; }
  else                { h[i] += c * (acc[i] + kv); }
}

// ---------------------------------------------------------------------------
// Pooling
// ---------------------------------------------------------------------------
__global__ void pool_cnt_kernel(const int* __restrict__ batch, float* __restrict__ cnt, int N) {
  int i = blockIdx.x * blockDim.x + threadIdx.x;
  if (i < N) atomicAdd(&cnt[batch[i]], 1.0f);
}
__global__ __launch_bounds__(256)
void pool_sum_kernel(const float* __restrict__ hout, const int* __restrict__ batch,
                     float* __restrict__ pooled, int N)
{
  int node = blockIdx.x * 8 + (threadIdx.x >> 5);
  int lane = threadIdx.x & 31;
  if (node >= N) return;
  int g = batch[node];
  const float4 v = *(const float4*)(hout + (size_t)node * HIDN + lane * 4);
  float* pr = pooled + (size_t)g * HIDN + lane * 4;
  atomicAdd(pr + 0, v.x); atomicAdd(pr + 1, v.y);
  atomicAdd(pr + 2, v.z); atomicAdd(pr + 3, v.w);
}
__global__ void pool_final_kernel(const float* __restrict__ pooled, const float* __restrict__ cnt,
                                  float* __restrict__ out) {
  int i = blockIdx.x * blockDim.x + threadIdx.x;
  if (i < NGRAPH * HIDN) out[i] = pooled[i] / fmaxf(cnt[i >> 7], 1.0f);
}

// ---------------------------------------------------------------------------
// Host orchestration
// ---------------------------------------------------------------------------
extern "C" void kernel_launch(void* const* d_in, const int* in_sizes, int n_in,
                              void* d_out, int out_size, void* d_ws, size_t ws_size,
                              hipStream_t stream)
{
  // inputs in setup_inputs() dict order
  const float* x       = (const float*)d_in[0];
  const int*   eidx    = (const int*)d_in[1];
  const int*   batch   = (const int*)d_in[2];
  const float* alpha   = (const float*)d_in[3];
  const float* beta    = (const float*)d_in[4];
  const float* phase   = (const float*)d_in[5];
  const float* W_in    = (const float*)d_in[6];
  const float* b_in    = (const float*)d_in[7];
  const float* g_in    = (const float*)d_in[8];
  const float* be_in   = (const float*)d_in[9];
  const float* W_alt   = (const float*)d_in[10];
  const float* b_alt   = (const float*)d_in[11];
  const float* g_alt   = (const float*)d_in[12];
  const float* be_alt  = (const float*)d_in[13];
  const float* W_self  = (const float*)d_in[14];
  const float* W_msg   = (const float*)d_in[15];
  const float* b_ode   = (const float*)d_in[16];
  const float* W_int   = (const float*)d_in[17];
  const float* b_int   = (const float*)d_in[18];
  const float* g_int   = (const float*)d_in[19];
  const float* be_int  = (const float*)d_in[20];
  const float* W_out   = (const float*)d_in[21];
  const float* b_out   = (const float*)d_in[22];
  float* out = (float*)d_out;

  const int N = in_sizes[0] / HIDN;      // 50000 nodes
  const int E = in_sizes[1] / 2;         // 800000 edges
  const int* src = eidx;
  const int* dst = eidx + E;

  // workspace layout
  const size_t NH = (size_t)N * HIDN;
  float* y0    = (float*)d_ws;           // scratch / tmp
  float* y1    = y0 + NH;                // scratch / kbuf / hout
  float* h     = y1 + NH;                // ODE state
  float* acc   = h + NH;                 // RK4 accumulator
  float* agg   = acc + NH;               // edge aggregation
  float* rdeg  = agg + NH;               // N floats (1/deg)
  float* cnt   = rdeg + (((size_t)N + 1023) & ~(size_t)1023);   // 64
  float* pooled = cnt + NGRAPH;          // 64*128
  // bf16 pre-swizzled weights (32 KB each, 32B aligned)
  unsigned short* wz = (unsigned short*)(pooled + NGRAPH * HIDN);
  unsigned short* wz_in   = wz;
  unsigned short* wz_alt  = wz + 1 * HIDN * HIDN;
  unsigned short* wz_self = wz + 2 * HIDN * HIDN;
  unsigned short* wz_msg  = wz + 3 * HIDN * HIDN;
  unsigned short* wz_eff  = wz + 4 * HIDN * HIDN;
  unsigned short* wz_out  = wz + 5 * HIDN * HIDN;

  const int gemm_blocks = (N + BM - 1) / BM;
  const int row_blocks  = (N + 7) / 8;       // wave-per-row kernels
  const int ew_blocks   = (int)((NH + 255) / 256);
  const int wz_blocks   = (HIDN * HIDN + 255) / 256;
  const float dt = 1.0f / 8.0f;              // DEPTH / ODE_STEPS

  // ---- one-time weight swizzles (bf16, B-fragment order) ----
  weight_swizzle_kernel<<<wz_blocks, 256, 0, stream>>>(W_in,  nullptr, wz_in);
  weight_swizzle_kernel<<<wz_blocks, 256, 0, stream>>>(W_alt, nullptr, wz_alt);
  weight_swizzle_kernel<<<wz_blocks, 256, 0, stream>>>(W_self, nullptr, wz_self);
  weight_swizzle_kernel<<<wz_blocks, 256, 0, stream>>>(W_msg, nullptr, wz_msg);
  weight_swizzle_kernel<<<wz_blocks, 256, 0, stream>>>(W_int, W_int + HIDN * HIDN, wz_eff);
  weight_swizzle_kernel<<<wz_blocks, 256, 0, stream>>>(W_out, nullptr, wz_out);

  // ---- degrees ----
  hipMemsetAsync(rdeg, 0, (size_t)N * sizeof(float), stream);
  deg_kernel<<<(E + 255) / 256, 256, 0, stream>>>(dst, rdeg, E);
  rdeg_finalize_kernel<<<(N + 255) / 256, 256, 0, stream>>>(rdeg, N);

  // ---- input superposition: h = s0*gelu(LN(x@W_in+b)) + s1*gelu(LN(x@W_alt+b)) ----
  gemm_kernel<false, false><<<gemm_blocks, 256, 0, stream>>>(
      x, wz_in, nullptr, nullptr, b_in, y0, N);
  gemm_kernel<false, false><<<gemm_blocks, 256, 0, stream>>>(
      x, wz_alt, nullptr, nullptr, b_alt, y1, N);
  ln_gelu_combine_kernel<<<row_blocks, 256, 0, stream>>>(
      y0, y1, g_in, be_in, g_alt, be_alt, alpha, beta, h, N);

  // ---- RK4: 8 steps, 4 ode_f evals each ----
  float* tmp  = y0;
  float* kbuf = y1;
  for (int step = 0; step < 8; ++step) {
    const float cs[3] = {0.5f * dt, 0.5f * dt, dt};
    for (int sub = 0; sub < 4; ++sub) {
      const float* fin = (sub == 0) ? h : tmp;
      // ode_f(fin) -> kbuf: agg = segsum(fin[src]*rdeg[dst]); tanh(fin@Wself+agg@Wmsg+b)
      hipMemsetAsync(agg, 0, NH * sizeof(float), stream);
      edge_agg_kernel<<<4096, 256, 0, stream>>>(fin, rdeg, src, dst, agg, E);
      gemm_kernel<true, true><<<gemm_blocks, 256, 0, stream>>>(
          fin, wz_self, agg, wz_msg, b_ode, kbuf, N);
      if (sub < 3)
        rk4_ew_kernel<<<ew_blocks, 256, 0, stream>>>(
            h, tmp, acc, kbuf, cs[sub], (sub == 0) ? 0 : 1, (int)NH);
      else
        rk4_ew_kernel<<<ew_blocks, 256, 0, stream>>>(
            h, tmp, acc, kbuf, dt / 6.0f, 2, (int)NH);
    }
  }

  // ---- interference mixing + collapse ----
  gemm_kernel<false, false><<<gemm_blocks, 256, 0, stream>>>(
      h, wz_eff, nullptr, nullptr, b_int, y0, N);
  interference_kernel<<<row_blocks, 256, 0, stream>>>(y0, g_int, be_int, alpha, beta, phase, h, N);

  // ---- output projection + global mean pool ----
  gemm_kernel<false, false><<<gemm_blocks, 256, 0, stream>>>(
      h, wz_out, nullptr, nullptr, b_out, y1, N);
  hipMemsetAsync(cnt, 0, NGRAPH * sizeof(float), stream);
  hipMemsetAsync(pooled, 0, (size_t)NGRAPH * HIDN * sizeof(float), stream);
  pool_cnt_kernel<<<(N + 255) / 256, 256, 0, stream>>>(batch, cnt, N);
  pool_sum_kernel<<<row_blocks, 256, 0, stream>>>(y1, batch, pooled, N);
  pool_final_kernel<<<(NGRAPH * HIDN + 255) / 256, 256, 0, stream>>>(pooled, cnt, out);
}
